// BiLSTM_72378788872375
// MI455X (gfx1250) — compile-verified, matrix operand order checked by reading, not run on hardware
//
#include <hip/hip_runtime.h>
#include <hip/hip_bf16.h>
#include <hip/hip_fp16.h>

typedef _Float16 f16;
typedef __attribute__((ext_vector_type(8)))  _Float16 v8h;
typedef __attribute__((ext_vector_type(16))) _Float16 v16h;
typedef __attribute__((ext_vector_type(8)))  float    v8f;

#define T_STEPS 512
#define BATCH   64
#define DIN     512
#define HID     512
#define G4      2048      // 4*H gate columns
#define KTOT    1024      // D + H fused K
#define NWG_DIR 16        // workgroups per direction
#define THREADS 256       // 8 wave32 per workgroup
#define LROWS   128       // weight rows resident in LDS per WG (2 h-tiles x 4 gates x 16)
#define LPITCH  1032      // halves per LDS row (1024 + 8 pad -> bank-conflict-free fragments)
#define LDS_BYTES ((size_t)LROWS * LPITCH * 2)   // 264,192 B of 320 KB

// ---- workspace layout (bytes) ----
constexpr size_t OFF_X16  = 0;                                          // x as f16 [T][B][D]
constexpr size_t OFF_WT   = OFF_X16 + (size_t)T_STEPS*BATCH*DIN*2;      // Wt f16 [2][G4][KTOT]
constexpr size_t OFF_BIAS = OFF_WT  + (size_t)2*G4*KTOT*2;              // bias f32 [2][G4]
constexpr size_t OFF_HBUF = OFF_BIAS + (size_t)2*G4*4;                  // h state f16 [2][2][B][H]
constexpr size_t OFF_CTR  = OFF_HBUF + (size_t)2*2*BATCH*HID*2;         // arrive counters [2]

__device__ __forceinline__ float sigm(float x)      { return 1.0f / (1.0f + __expf(-x)); }
__device__ __forceinline__ float tanh_fast(float x) { return 2.0f * sigm(2.0f * x) - 1.0f; }

// assemble a lane's 16-half fragment from two contiguous 16B chunks (P, P+16 halves)
#define LD16(P) ({ v16h _a; *(v8h*)&_a = *(const v8h*)(P); ((v8h*)&_a)[1] = *(const v8h*)((P) + 16); _a; })

__global__ void init_ws(unsigned* p, int n) {
    int i = blockIdx.x * blockDim.x + threadIdx.x;
    if (i < n) p[i] = 0u;
}

__global__ void cvt_x(const float* __restrict__ x, f16* __restrict__ x16, int n) {
    int stride = gridDim.x * blockDim.x;
    for (int i = blockIdx.x * blockDim.x + threadIdx.x; i < n; i += stride)
        x16[i] = (f16)x[i];
}

// Wt[n][k] = (k < D ? Wih[k][n] : Whh[k-D][n]) as f16  -> column-major weights for B-operand
__global__ void cvt_w(const float* __restrict__ Wih, const float* __restrict__ Whh,
                      f16* __restrict__ Wt) {
    int idx = blockIdx.x * blockDim.x + threadIdx.x;
    if (idx >= G4 * KTOT) return;
    int n = idx / KTOT, k = idx % KTOT;
    float v = (k < DIN) ? Wih[(size_t)k * G4 + n] : Whh[(size_t)(k - DIN) * G4 + n];
    Wt[idx] = (f16)v;
}

__global__ void cvt_b(const float* __restrict__ bih, const float* __restrict__ bhh,
                      float* __restrict__ bsum) {
    int i = blockIdx.x * blockDim.x + threadIdx.x;
    if (i < G4) bsum[i] = bih[i] + bhh[i];
}

__global__ __launch_bounds__(THREADS, 1)
void bilstm_scan(const f16* __restrict__ x16,       // [T][B][D] f16
                 const f16* __restrict__ WtAll,     // [2][G4][KTOT] f16
                 const float* __restrict__ biasAll, // [2][G4]
                 f16* __restrict__ hbufAll,         // [2][2][B][H] f16 (double buffered)
                 unsigned* __restrict__ ctrAll,     // [2] arrive counters
                 float* __restrict__ out)           // h [T][B][2H] ++ h_n [B][2H] ++ c_n [B][2H]
{
    extern __shared__ f16 lds[];                    // WG's weight slice, 512x-reused

    const int wgid = blockIdx.x;
    const int dir  = wgid / NWG_DIR;
    const int wg   = wgid % NWG_DIR;
    const int tid  = threadIdx.x;
    const int lane = tid & 31;
    const int wave = tid >> 5;            // 0..7
    const int m    = wave & 3;            // M tile: batch rows [m*16, m*16+16)
    const int jl   = wave >> 2;           // local h tile 0..1
    const int jt   = wg * 2 + jl;         // global h column tile 0..31

    const f16*   Wt   = WtAll   + (size_t)dir * G4 * KTOT;
    const float* bias = biasAll + (size_t)dir * G4;
    f16* hb0 = hbufAll + (size_t)dir * 2 * BATCH * HID;
    unsigned* myctr = ctrAll + dir;

    // ---- stage this WG's 128 weight rows (2 h-tiles x 4 gates) into LDS ----
    // row rl = (g*2 + j)*16 + nl  maps to global gate row n = g*HID + (wg*2 + j)*16 + nl
    for (int i = tid; i < LROWS * (KTOT / 8); i += THREADS) {
        const int rl = i >> 7;            // LDS row
        const int ck = i & 127;           // 16B chunk within row
        const int g  = rl >> 5;
        const int j  = (rl >> 4) & 1;
        const int nl = rl & 15;
        const int n  = g * HID + (wg * 2 + j) * 16 + nl;
        *(v8h*)(lds + (size_t)rl * LPITCH + ck * 8) =
            *((const v8h*)(Wt + (size_t)n * KTOT) + ck);
    }
    __syncthreads();

    const int nlane = lane & 15;
    const int lsel  = (lane >> 4) & 1;    // lane-half selects K sub-chunk
    const int arow  = m * 16 + nlane;     // A-operand batch row for this lane

    // per-gate LDS row bases + fused bias (constant across timesteps)
    const f16* l0 = lds + (size_t)((0 * 2 + jl) * 16 + nlane) * LPITCH;
    const f16* l1 = lds + (size_t)((1 * 2 + jl) * 16 + nlane) * LPITCH;
    const f16* l2 = lds + (size_t)((2 * 2 + jl) * 16 + nlane) * LPITCH;
    const f16* l3 = lds + (size_t)((3 * 2 + jl) * 16 + nlane) * LPITCH;
    const float bi = bias[0 * HID + jt * 16 + nlane];
    const float bf = bias[1 * HID + jt * 16 + nlane];
    const float bg = bias[2 * HID + jt * 16 + nlane];
    const float bo = bias[3 * HID + jt * 16 + nlane];

    v8f c = {0.f, 0.f, 0.f, 0.f, 0.f, 0.f, 0.f, 0.f};  // cell state, register-resident

    const size_t HN_OFF = (size_t)T_STEPS * BATCH * 2 * HID;
    const size_t CN_OFF = HN_OFF + (size_t)BATCH * 2 * HID;

    for (int t = 0; t < T_STEPS; ++t) {
        // stop LLVM from hoisting the (loop-invariant) LDS weight fragments out of
        // the t-loop and spilling them to scratch: force reload-per-step (cheap ds ops)
        asm volatile("" ::: "memory");

        const int p = t & 1;
        const f16* hcur  = hb0 + (size_t)p * BATCH * HID;        // read state(t)
        f16*       hnext = hb0 + (size_t)(1 - p) * BATCH * HID;  // write state(t+1)
        const f16* xa = x16 + ((size_t)t * BATCH + arow) * DIN;
        const f16* ha = hcur + (size_t)arow * HID;

        __builtin_prefetch(xa + (size_t)BATCH * DIN, 0, 0);  // warm next timestep's x

        v8f a0 = {0.f,0.f,0.f,0.f,0.f,0.f,0.f,0.f};
        v8f a1 = a0, a2 = a0, a3 = a0;

        // K in [0,512): input projection x_t @ Wih
        #pragma unroll
        for (int kc = 0; kc < 16; ++kc) {
            const int kb = kc * 32 + lsel * 8;
            v16h A  = LD16(xa + kb);
            v16h B0 = LD16(l0 + kb);
            a0 = __builtin_amdgcn_wmma_f32_16x16x32_f16(false, A, false, B0, (short)0, a0, false, false);
            v16h B1 = LD16(l1 + kb);
            a1 = __builtin_amdgcn_wmma_f32_16x16x32_f16(false, A, false, B1, (short)0, a1, false, false);
            v16h B2 = LD16(l2 + kb);
            a2 = __builtin_amdgcn_wmma_f32_16x16x32_f16(false, A, false, B2, (short)0, a2, false, false);
            v16h B3 = LD16(l3 + kb);
            a3 = __builtin_amdgcn_wmma_f32_16x16x32_f16(false, A, false, B3, (short)0, a3, false, false);
        }
        // K in [512,1024): recurrence h @ Whh
        #pragma unroll
        for (int kc = 0; kc < 16; ++kc) {
            const int kb = kc * 32 + lsel * 8;
            const int kw = DIN + kb;
            v16h A  = LD16(ha + kb);
            v16h B0 = LD16(l0 + kw);
            a0 = __builtin_amdgcn_wmma_f32_16x16x32_f16(false, A, false, B0, (short)0, a0, false, false);
            v16h B1 = LD16(l1 + kw);
            a1 = __builtin_amdgcn_wmma_f32_16x16x32_f16(false, A, false, B1, (short)0, a1, false, false);
            v16h B2 = LD16(l2 + kw);
            a2 = __builtin_amdgcn_wmma_f32_16x16x32_f16(false, A, false, B2, (short)0, a2, false, false);
            v16h B3 = LD16(l3 + kw);
            a3 = __builtin_amdgcn_wmma_f32_16x16x32_f16(false, A, false, B3, (short)0, a3, false, false);
        }

        // gate nonlinearity: i/f/g/o tiles share lane/VGPR coordinates -> pure VALU
        const int rbase = m * 16 + lsel * 8;   // C/D layout: lanes 16-31 hold M+8
        const int hcol  = jt * 16 + nlane;
        #pragma unroll
        for (int v = 0; v < 8; ++v) {
            const int row = rbase + v;
            float ig = sigm(a0[v] + bi);
            float fg = sigm(a1[v] + bf);
            float gg = tanh_fast(a2[v] + bg);
            float og = sigm(a3[v] + bo);
            float cn = fg * c[v] + ig * gg;
            float hn = og * tanh_fast(cn);
            c[v] = cn;
            hnext[(size_t)row * HID + hcol] = (f16)hn;  // state for next step (L2)
            const size_t oidx = ((size_t)t * BATCH + row) * (2 * HID) + (size_t)dir * HID + hcol;
            __builtin_nontemporal_store(hn, &out[oidx]); // 134MB stream: keep L2 for weights
            if (t == T_STEPS - 1) {
                const size_t fidx = (size_t)row * (2 * HID) + (size_t)dir * HID + hcol;
                out[HN_OFF + fidx] = hn;
                out[CN_OFF + fidx] = cn;
            }
        }

        // per-direction device-scope split barrier: release stores, arrive, spin-acquire
        __threadfence();
        __syncthreads();
        if (tid == 0) {
            __hip_atomic_fetch_add(myctr, 1u, __ATOMIC_RELEASE, __HIP_MEMORY_SCOPE_AGENT);
            const unsigned target = (unsigned)(t + 1) * NWG_DIR;
            while (__hip_atomic_load(myctr, __ATOMIC_ACQUIRE, __HIP_MEMORY_SCOPE_AGENT) < target)
                __builtin_amdgcn_s_sleep(1);
        }
        __syncthreads();
    }
}

extern "C" void kernel_launch(void* const* d_in, const int* in_sizes, int n_in,
                              void* d_out, int out_size, void* d_ws, size_t ws_size,
                              hipStream_t stream) {
    (void)in_sizes; (void)n_in; (void)out_size; (void)ws_size;
    const float* x     = (const float*)d_in[0];
    const float* Wih_f = (const float*)d_in[1];
    const float* Whh_f = (const float*)d_in[2];
    const float* bih_f = (const float*)d_in[3];
    const float* bhh_f = (const float*)d_in[4];
    const float* Wih_b = (const float*)d_in[5];
    const float* Whh_b = (const float*)d_in[6];
    const float* bih_b = (const float*)d_in[7];
    const float* bhh_b = (const float*)d_in[8];

    char* ws = (char*)d_ws;
    f16*      x16  = (f16*)(ws + OFF_X16);
    f16*      Wt   = (f16*)(ws + OFF_WT);
    float*    bias = (float*)(ws + OFF_BIAS);
    f16*      hbuf = (f16*)(ws + OFF_HBUF);
    unsigned* ctr  = (unsigned*)(ws + OFF_CTR);

    // zero h-state double buffers + arrive counters every call (determinism)
    const int nzero = (int)((OFF_CTR + 256 - OFF_HBUF) / 4);
    init_ws<<<(nzero + 255) / 256, 256, 0, stream>>>((unsigned*)(ws + OFF_HBUF), nzero);

    const int nx = T_STEPS * BATCH * DIN;
    cvt_x<<<8192, 256, 0, stream>>>(x, x16, nx);

    const int nw = G4 * KTOT;
    cvt_w<<<(nw + 255) / 256, 256, 0, stream>>>(Wih_f, Whh_f, Wt);
    cvt_w<<<(nw + 255) / 256, 256, 0, stream>>>(Wih_b, Whh_b, Wt + (size_t)G4 * KTOT);
    cvt_b<<<(G4 + 255) / 256, 256, 0, stream>>>(bih_f, bhh_f, bias);
    cvt_b<<<(G4 + 255) / 256, 256, 0, stream>>>(bih_b, bhh_b, bias + G4);

    bilstm_scan<<<2 * NWG_DIR, THREADS, LDS_BYTES, stream>>>(x16, Wt, bias, hbuf, ctr, (float*)d_out);
}
